// CrossNet_27118423507311
// MI455X (gfx1250) — matrix operand order repaired and checked by reading, not compiled, not run on hardware
//
#include <hip/hip_runtime.h>
#include <math.h>

#define B_N 4096
#define T_N 128
#define F_N 96
#define S_N 128
#define H_N 512
#define GN  (4*H_N)    // 2048 gate width
#define IN2 (2*F_N)    // 192 lstm input width
#define IWGT 0.3f
#define L1C 0.01f
#define L2C 0.1f

typedef __attribute__((ext_vector_type(16))) __bf16        v16bf;
typedef __attribute__((ext_vector_type(8)))  float         v8f;
typedef __attribute__((ext_vector_type(4)))  unsigned int  u32x4;

// global (addrspace 1) flavors so weight loads lower to global_load_b128
// (LOADcnt only, SADDR+voffset form) instead of flat_load (LOADcnt+DScnt).
typedef __attribute__((address_space(1))) const u32x4 g_u32x4;

union FragU { u32x4 u[2]; v16bf v; };

__device__ __forceinline__ v8f wmma_bf(v16bf a, v16bf b, v8f c) {
  return __builtin_amdgcn_wmma_f32_16x16x32_bf16(false, a, false, b, (short)0, c, false, false);
}

__device__ __forceinline__ v8f vzero8() {
  v8f z;
#pragma unroll
  for (int i = 0; i < 8; ++i) z[i] = 0.f;
  return z;
}

// A fragment (16x32 bf16) from LDS: lane ln = row m;
// elements 0..7 = K kc..kc+7, 8..15 = K kc+16..kc+23 (caller passes kc = k0 + hi*8)
__device__ __forceinline__ v16bf fragA(const __bf16* base, int ld, int m, int kc) {
  const __bf16* p = base + m * ld + kc;
  FragU f;
  f.u[0] = *(const u32x4*)p;
  f.u[1] = *(const u32x4*)(p + 16);
  return f.v;
}

// B fragment (32x16 bf16) from global weights: lane ln = col n (row n of W).
// elements 0..7 = K kc..kc+7, 8..15 = K kc+16..kc+23 (kc = k0 + hi*8). Cached global loads.
__device__ __forceinline__ v16bf fragB(const __bf16* base, int ld, int n, int kc) {
  const __bf16* p = base + n * ld + kc;
  FragU f;
  f.u[0] = *(const g_u32x4*)p;
  f.u[1] = *(const g_u32x4*)(p + 16);
  return f.v;
}

__device__ __forceinline__ float sigm(float x) { return 1.f / (1.f + __expf(-x)); }

// ---------------------------------------------------------------------------
// Weight fp32 -> bf16 conversion + accumulator zeroing
// ---------------------------------------------------------------------------
__global__ __launch_bounds__(256)
void init_convert_kernel(const float* td, const float* hist, const float* real,
                         const float* wih, const float* whh,
                         __bf16* tdb, __bf16* histb, __bf16* realb,
                         __bf16* wihb, __bf16* whhb,
                         float* accum, int accum_n) {
  const int n_td = H_N * F_N, n_hist = F_N * H_N, n_real = F_N * F_N;
  const int n_wih = GN * IN2, n_whh = GN * H_N;
  const int total = n_td + n_hist + n_real + n_wih + n_whh;
  int g0 = blockIdx.x * blockDim.x + threadIdx.x;
  for (int i = g0; i < total; i += gridDim.x * blockDim.x) {
    int j = i;
    if (j < n_td)                 { tdb[j]   = (__bf16)td[j]; }
    else if ((j -= n_td) < n_hist){ histb[j] = (__bf16)hist[j]; }
    else if ((j -= n_hist) < n_real){ realb[j] = (__bf16)real[j]; }
    else if ((j -= n_real) < n_wih){ wihb[j]  = (__bf16)wih[j]; }
    else { j -= n_wih; whhb[j] = (__bf16)whh[j]; }
  }
  if (g0 < accum_n) accum[g0] = 0.f;
}

// ---------------------------------------------------------------------------
// Static reconstruction: s_h = statics@fr_W.T + fr_b ; s_c ; s_loss partials
// ---------------------------------------------------------------------------
__global__ __launch_bounds__(256)
void static_recon_kernel(const float* __restrict__ statics, const float* __restrict__ smask,
                         const float* __restrict__ frW, const float* __restrict__ frb,
                         float* __restrict__ s_c, float* __restrict__ sloss_acc) {
  __shared__ float red[2];
  if (threadIdx.x == 0) { red[0] = 0.f; red[1] = 0.f; }
  __syncthreads();
  int idx = blockIdx.x * 256 + threadIdx.x;       // over B*S
  int b = idx >> 7, j = idx & (S_N - 1);
  const float* sr = statics + (size_t)b * S_N;
  const float* wr = frW + (size_t)j * S_N;
  float acc = frb[j];
  for (int k = 0; k < S_N; ++k) acc += sr[k] * wr[k];
  float sm = smask[idx], st = statics[idx];
  float sc = sm * st + (1.f - sm) * acc;
  s_c[idx] = sc;
  atomicAdd(&red[0], fabsf(st - sc) * sm);
  atomicAdd(&red[1], sm);
  __syncthreads();
  if (threadIdx.x == 0) {
    atomicAdd(&sloss_acc[0], red[0]);
    atomicAdd(&sloss_acc[1], red[1]);
  }
}

// ---------------------------------------------------------------------------
// Generic dense: out[b,n] = (relu?)(in[b,:]@W[n,:] + bias[n])  (one-time cost)
// ---------------------------------------------------------------------------
__global__ __launch_bounds__(256)
void dense_kernel(const float* __restrict__ in, const float* __restrict__ W,
                  const float* __restrict__ bias, float* __restrict__ out,
                  int K, int N, int do_relu) {
  int idx = blockIdx.x * 256 + threadIdx.x;
  int b = idx / N, n = idx - b * N;
  const float* ir = in + (size_t)b * K;
  const float* wr = W + (size_t)n * K;
  float acc = bias[n];
  for (int k = 0; k < K; ++k) acc += ir[k] * wr[k];
  if (do_relu) acc = fmaxf(acc, 0.f);
  out[idx] = acc;
}

// ---------------------------------------------------------------------------
// Main recurrent kernel: 16 batch rows per block, all T steps in-kernel.
// 8 waves; each wave owns a 64-column slice of H (h, c live in VGPRs).
// ---------------------------------------------------------------------------
__global__ __launch_bounds__(256)
void scan_kernel(const float* __restrict__ values, const float* __restrict__ masks,
                 const float* __restrict__ deltas, const float* __restrict__ labels,
                 const float* __restrict__ td_b, const float* __restrict__ hist_b,
                 const float* __restrict__ real_b, const float* __restrict__ bih,
                 const float* __restrict__ bhh, const float* __restrict__ outW,
                 const float* __restrict__ outb,
                 const __bf16* __restrict__ tdW, const __bf16* __restrict__ histW,
                 const __bf16* __restrict__ realW, const __bf16* __restrict__ Wih,
                 const __bf16* __restrict__ Whh,
                 const float* __restrict__ h0, const float* __restrict__ stterm,
                 float* __restrict__ out_sig, float* __restrict__ out_imp,
                 float* __restrict__ num, float* __restrict__ msum,
                 float* __restrict__ yloss) {
  __shared__ __bf16 sh_h[16 * H_N];     // decayed h (A operand), bf16
  __shared__ __bf16 sh_inp[16 * IN2];   // [c_c | m] (A operand for Wih)
  __shared__ __bf16 sh_x[16 * F_N];     // x bf16 (A for real_W)
  __shared__ __bf16 sh_d[16 * F_N];     // delta bf16 (A for td_W)
  __shared__ float  sh_red[16];

  const int tid  = threadIdx.x;
  const int lane = tid & 31;
  const int wave = tid >> 5;
  const int ln   = lane & 15;
  const int hi   = lane >> 4;
  const int kA   = hi * 8;      // A-fragment K chunk base
  const int kB   = hi * 8;      // B-fragment K chunk base
  const int b0   = blockIdx.x * 16;
  const int ncw  = wave * 64;   // this wave's h-column base

  // Loop-laundered weight pointers (see asm in loop body): prevents LICM from
  // hoisting loop-invariant weight fragments into registers (-> spills).
  const __bf16* tdW_l   = tdW;
  const __bf16* histW_l = histW;
  const __bf16* realW_l = realW;
  const __bf16* Wih_l   = Wih;
  const __bf16* Whh_l   = Whh;

  // Per-wave register-resident state: h, c (D-layout: row = r + hi*8, col = ncw+t*16+ln)
  float h[4][8], c[4][8];
  float tdb[4], gb[4][4];
#pragma unroll
  for (int t = 0; t < 4; ++t) {
    int col = ncw + t * 16 + ln;
    tdb[t] = td_b[col];
#pragma unroll
    for (int q = 0; q < 4; ++q) gb[t][q] = bih[q * H_N + col] + bhh[q * H_N + col];
#pragma unroll
    for (int r = 0; r < 8; ++r) {
      h[t][r] = h0[(size_t)(b0 + r + hi * 8) * H_N + col];
      c[t][r] = 0.f;
    }
  }
  float xcb = 0.f, stv[8];
#pragma unroll
  for (int r = 0; r < 8; ++r) stv[r] = 0.f;
  if (wave < 6) {
    int nB = wave * 16 + ln;
    xcb = real_b[nB] + hist_b[nB];
#pragma unroll
    for (int r = 0; r < 8; ++r) stv[r] = stterm[(size_t)(b0 + r + hi * 8) * F_N + nB];
  }

  for (int ts = 0; ts < T_N; ++ts) {
    // Launder weight pointers: their defs now live inside the loop, so weight
    // fragment loads cannot be hoisted across iterations. Loads stay ordinary
    // cached global_load_b128 (L2 hits) with compiler-managed waits.
    asm volatile("" : "+s"(tdW_l), "+s"(histW_l), "+s"(realW_l),
                      "+s"(Wih_l), "+s"(Whh_l));

    // ---- stage x, delta, mask into LDS (bf16) ----
    for (int e = tid; e < 16 * F_N; e += 256) {
      int row = e / F_N, col = e - row * F_N;
      size_t g = ((size_t)(b0 + row) * T_N + ts) * F_N + col;
      sh_x[row * F_N + col] = (__bf16)values[g];
      sh_d[row * F_N + col] = (__bf16)deltas[g];
      sh_inp[row * IN2 + F_N + col] = (__bf16)masks[g];
    }
    if (tid == 0) { sh_red[0] = 0.f; sh_red[1] = 0.f; }
    __syncthreads();

    // ---- Phase A: gamma = exp(-relu(delta@tdW.T + td_b)); decay h; stage bf16 h ----
#pragma unroll
    for (int t = 0; t < 4; ++t) {
      v8f acc = vzero8();
      int n0 = ncw + t * 16;
#pragma unroll
      for (int k = 0; k < 3; ++k) {
        v16bf a  = fragA(sh_d, F_N, ln, k * 32 + kA);
        v16bf bw = fragB(tdW_l,  F_N, n0 + ln, k * 32 + kB);
        acc = wmma_bf(a, bw, acc);
      }
#pragma unroll
      for (int r = 0; r < 8; ++r) {
        float gmm = __expf(-fmaxf(acc[r] + tdb[t], 0.f));
        h[t][r] *= gmm;
        sh_h[(r + hi * 8) * H_N + n0 + ln] = (__bf16)h[t][r];
      }
    }
    __syncthreads();

    // ---- Phase B: x_c = x@realW.T + h@histW.T + biases + st_term; c_c; losses ----
    if (wave < 6) {
      v8f acc = vzero8();
      int nB0 = wave * 16;
#pragma unroll
      for (int k = 0; k < 3; ++k) {
        v16bf a  = fragA(sh_x, F_N, ln, k * 32 + kA);
        v16bf bw = fragB(realW_l, F_N, nB0 + ln, k * 32 + kB);
        acc = wmma_bf(a, bw, acc);
      }
      for (int k = 0; k < 16; ++k) {
        v16bf a  = fragA(sh_h, H_N, ln, k * 32 + kA);
        v16bf bw = fragB(histW_l, H_N, nB0 + ln, k * 32 + kB);
        acc = wmma_bf(a, bw, acc);
      }
      float ls = 0.f, lm = 0.f;
#pragma unroll
      for (int r = 0; r < 8; ++r) {
        int row = r + hi * 8;
        size_t g = ((size_t)(b0 + row) * T_N + ts) * F_N + nB0 + ln;
        float xv = values[g], mv = masks[g];
        float xc = acc[r] + xcb + stv[r];
        ls += fabsf(xv - xc) * mv;
        lm += mv;
        float cc = mv * xv + (1.f - mv) * xc;
        out_imp[g] = cc;
        sh_inp[row * IN2 + nB0 + ln] = (__bf16)cc;
      }
      atomicAdd(&sh_red[0], ls);
      atomicAdd(&sh_red[1], lm);
    }
    __syncthreads();

    if (tid == 0) {
      atomicAdd(&num[ts],  sh_red[0]);
      atomicAdd(&msum[ts], sh_red[1]);
    }

    // ---- Phase C: gates = inp@Wih.T + h@Whh.T + biases; LSTM update ----
#pragma unroll
    for (int t = 0; t < 4; ++t) {
      v8f acc[4];
#pragma unroll
      for (int q = 0; q < 4; ++q) acc[q] = vzero8();
      int n0 = ncw + t * 16 + ln;
      for (int k = 0; k < 6; ++k) {
        v16bf a = fragA(sh_inp, IN2, ln, k * 32 + kA);
#pragma unroll
        for (int q = 0; q < 4; ++q) {
          v16bf bw = fragB(Wih_l, IN2, q * H_N + n0, k * 32 + kB);
          acc[q] = wmma_bf(a, bw, acc[q]);
        }
      }
      for (int k = 0; k < 16; ++k) {
        v16bf a = fragA(sh_h, H_N, ln, k * 32 + kA);
#pragma unroll
        for (int q = 0; q < 4; ++q) {
          v16bf bw = fragB(Whh_l, H_N, q * H_N + n0, k * 32 + kB);
          acc[q] = wmma_bf(a, bw, acc[q]);
        }
      }
#pragma unroll
      for (int r = 0; r < 8; ++r) {
        float iv = sigm(acc[0][r] + gb[t][0]);
        float fv = sigm(acc[1][r] + gb[t][1]);
        float gv = tanhf(acc[2][r] + gb[t][2]);
        float ov = sigm(acc[3][r] + gb[t][3]);
        float cn = fv * c[t][r] + iv * gv;
        c[t][r] = cn;
        h[t][r] = ov * tanhf(cn);
      }
    }
    __syncthreads();
  }

  // ---- output head: z = h@out_W.T + out_b; sigmoid; BCE-with-logits ----
  if (tid < 16) sh_red[tid] = 0.f;
  __syncthreads();
  {
    float ow[4];
#pragma unroll
    for (int t = 0; t < 4; ++t) ow[t] = outW[ncw + t * 16 + ln];
#pragma unroll
    for (int r = 0; r < 8; ++r) {
      float s = 0.f;
#pragma unroll
      for (int t = 0; t < 4; ++t) s += h[t][r] * ow[t];
      atomicAdd(&sh_red[r + hi * 8], s);
    }
  }
  __syncthreads();
  if (tid < 16) {
    float z = sh_red[tid] + outb[0];
    out_sig[b0 + tid] = 1.f / (1.f + __expf(-z));
    float lb = labels[b0 + tid];
    float bce = fmaxf(z, 0.f) - z * lb + log1pf(__expf(-fabsf(z)));
    atomicAdd(yloss, bce);
  }
}

// ---------------------------------------------------------------------------
// Regularizer sums (single block)
// rl: 0=sum|real| 1=sum|hist| 2=sum|st| 3=sum|diag real| 4=sum|diag fr|
// ---------------------------------------------------------------------------
__global__ __launch_bounds__(256)
void rloss_kernel(const float* __restrict__ realW, const float* __restrict__ histW,
                  const float* __restrict__ stW, const float* __restrict__ frW,
                  float* __restrict__ rl) {
  __shared__ float red[5];
  if (threadIdx.x < 5) red[threadIdx.x] = 0.f;
  __syncthreads();
  float a0 = 0, a1 = 0, a2 = 0, a3 = 0, a4 = 0;
  for (int i = threadIdx.x; i < F_N * F_N; i += 256) a0 += fabsf(realW[i]);
  for (int i = threadIdx.x; i < F_N * H_N; i += 256) a1 += fabsf(histW[i]);
  for (int i = threadIdx.x; i < F_N * S_N; i += 256) a2 += fabsf(stW[i]);
  for (int i = threadIdx.x; i < F_N; i += 256) a3 += fabsf(realW[i * F_N + i]);
  for (int i = threadIdx.x; i < S_N; i += 256) a4 += fabsf(frW[i * S_N + i]);
  atomicAdd(&red[0], a0);
  atomicAdd(&red[1], a1);
  atomicAdd(&red[2], a2);
  atomicAdd(&red[3], a3);
  atomicAdd(&red[4], a4);
  __syncthreads();
  if (threadIdx.x < 5) rl[threadIdx.x] = red[threadIdx.x];
}

// ---------------------------------------------------------------------------
// Final scalar loss
// ---------------------------------------------------------------------------
__global__ void final_kernel(const float* num, const float* msum, const float* yloss,
                             const float* sloss_acc, const float* rl, float* out_loss) {
  if (threadIdx.x == 0 && blockIdx.x == 0) {
    float xl = 0.f;
    for (int t = 0; t < T_N; ++t) xl += num[t] / (msum[t] + 1e-5f);
    float yl = yloss[0] / (float)B_N;
    float sl = sloss_acc[0] / (sloss_acc[1] + 1e-5f) + IWGT * L2C * rl[4];
    float rv = L1C * (rl[0] + rl[1] + rl[2]) + L2C * rl[3];
    out_loss[0] = yl + rv + IWGT * xl + sl;
  }
}

// ---------------------------------------------------------------------------
extern "C" void kernel_launch(void* const* d_in, const int* in_sizes, int n_in,
                              void* d_out, int out_size, void* d_ws, size_t ws_size,
                              hipStream_t stream) {
  (void)in_sizes; (void)n_in; (void)out_size; (void)ws_size;
  const float* values       = (const float*)d_in[0];
  const float* masks        = (const float*)d_in[1];
  const float* deltas       = (const float*)d_in[2];
  const float* statics      = (const float*)d_in[3];
  const float* static_masks = (const float*)d_in[4];
  const float* labels       = (const float*)d_in[5];
  const float* fr_W  = (const float*)d_in[6];
  const float* fr_b  = (const float*)d_in[7];
  const float* d1_W  = (const float*)d_in[8];
  const float* d1_b  = (const float*)d_in[9];
  const float* d2_W  = (const float*)d_in[10];
  const float* d2_b  = (const float*)d_in[11];
  const float* d3_W  = (const float*)d_in[12];
  const float* d3_b  = (const float*)d_in[13];
  const float* td_W  = (const float*)d_in[14];
  const float* td_b  = (const float*)d_in[15];
  const float* hist_W = (const float*)d_in[16];
  const float* hist_b = (const float*)d_in[17];
  const float* st_W  = (const float*)d_in[18];
  const float* st_b  = (const float*)d_in[19];
  const float* real_W = (const float*)d_in[20];
  const float* real_b = (const float*)d_in[21];
  const float* lstm_Wih = (const float*)d_in[22];
  const float* lstm_Whh = (const float*)d_in[23];
  const float* lstm_bih = (const float*)d_in[24];
  const float* lstm_bhh = (const float*)d_in[25];
  const float* out_W = (const float*)d_in[26];
  const float* out_b = (const float*)d_in[27];

  // ---- workspace layout ----
  char* ws = (char*)d_ws;
  size_t off = 0;
  __bf16* tdWb   = (__bf16*)(ws + off); off += (size_t)H_N * F_N * 2;
  __bf16* histWb = (__bf16*)(ws + off); off += (size_t)F_N * H_N * 2;
  __bf16* realWb = (__bf16*)(ws + off); off += (size_t)F_N * F_N * 2;
  __bf16* Wihb   = (__bf16*)(ws + off); off += (size_t)GN * IN2 * 2;
  __bf16* Whhb   = (__bf16*)(ws + off); off += (size_t)GN * H_N * 2;
  off = (off + 15) & ~(size_t)15;
  float* s_c    = (float*)(ws + off); off += (size_t)B_N * S_N * 4;
  float* stterm = (float*)(ws + off); off += (size_t)B_N * F_N * 4;
  float* hA     = (float*)(ws + off); off += (size_t)B_N * H_N * 4;
  float* accum  = (float*)(ws + off); off += 512 * 4;
  float* num_a  = accum;         // [128]
  float* msum_a = accum + 128;   // [128]
  float* yloss  = accum + 256;   // [1]
  float* sloss  = accum + 257;   // [2] num, den
  float* rl     = accum + 260;   // [5]

  float* outp     = (float*)d_out;
  float* out_loss = outp;
  float* out_sig  = outp + 1;
  float* out_imp  = outp + 1 + B_N;
  float* hB       = out_imp;     // temp scratch; fully overwritten by scan_kernel

  // 1) convert weights to bf16 + zero accumulators
  init_convert_kernel<<<7588, 256, 0, stream>>>(td_W, hist_W, real_W, lstm_Wih, lstm_Whh,
                                                tdWb, histWb, realWb, Wihb, Whhb,
                                                accum, 266);
  // 2) static reconstruction
  static_recon_kernel<<<(B_N * S_N) / 256, 256, 0, stream>>>(statics, static_masks, fr_W, fr_b,
                                                             s_c, sloss);
  // 3) dense stack s_c -> h0 (hA), plus st_term
  dense_kernel<<<(B_N * H_N) / 256, 256, 0, stream>>>(s_c, d1_W, d1_b, hA, S_N, H_N, 1);
  dense_kernel<<<(B_N * H_N) / 256, 256, 0, stream>>>(hA, d2_W, d2_b, hB, H_N, H_N, 1);
  dense_kernel<<<(B_N * H_N) / 256, 256, 0, stream>>>(hB, d3_W, d3_b, hA, H_N, H_N, 1);
  dense_kernel<<<(B_N * F_N) / 256, 256, 0, stream>>>(s_c, st_W, st_b, stterm, S_N, F_N, 0);
  // 4) recurrent scan (WMMA), writes sigmoid output + imputations + loss partials
  scan_kernel<<<B_N / 16, 256, 0, stream>>>(values, masks, deltas, labels,
                                            td_b, hist_b, real_b, lstm_bih, lstm_bhh,
                                            out_W, out_b,
                                            tdWb, histWb, realWb, Wihb, Whhb,
                                            hA, stterm,
                                            out_sig, out_imp, num_a, msum_a, yloss);
  // 5) regularizers + final loss scalar
  rloss_kernel<<<1, 256, 0, stream>>>(real_W, hist_W, st_W, fr_W, rl);
  final_kernel<<<1, 64, 0, stream>>>(num_a, msum_a, yloss, sloss, rl, out_loss);
}
